// ATDGNN_10204842295627
// MI455X (gfx1250) — compile-verified, matrix operand order checked by reading, not run on hardware
//
#include <hip/hip_runtime.h>
#include <hip/hip_bf16.h>
#include <math.h>

typedef __attribute__((ext_vector_type(16))) _Float16 v16h;
typedef __attribute__((ext_vector_type(8)))  float    v8f;
typedef __attribute__((ext_vector_type(2)))  float    v2f;
typedef __attribute__((ext_vector_type(4)))  unsigned u32x4;
typedef __attribute__((ext_vector_type(8)))  unsigned u32x8;

#define BATCH 32
#define LTOT  547
#define NW    23
#define WINL  100
#define SEQS  (BATCH*NW)      // 736
#define ROWS  (SEQS*WINL)     // 73600
#define MFI   (BATCH*LTOT)    // 17504
#define KFI   2048
#define EPSF  1e-5f

// ---------------------------------------------------------------------------
// Fused conv2d (1xK taps) + squared pooling + log + scatter into the f16 GEMM
// operand. One block per (b,f,c) row: input row + taps staged in LDS, each
// conv point computed ONCE into LDS y^2, then pooled (4x less VALU work than
// the per-output recompute version).
// ---------------------------------------------------------------------------
__global__ __launch_bounds__(256) void k_conv_power(const float* __restrict__ x,
                                                    const float* __restrict__ w,
                                                    const float* __restrict__ bias,
                                                    _Float16* __restrict__ X16,
                                                    int K, int Lc, int Lp, int off) {
  int bid = blockIdx.x;
  int c  = bid % 32;  int t2 = bid / 32;
  int f  = t2 % 64;   int b  = t2 / 64;
  __shared__ float xs[800];
  __shared__ float wsm[100];
  __shared__ float yy[800];
  int tid = threadIdx.x;
  const float* xr = x + ((size_t)b * 32 + c) * 800;
  for (int i = tid; i < 800; i += 256) xs[i] = xr[i];
  for (int i = tid; i < K; i += 256) wsm[i] = w[(size_t)f * K + i];
  __syncthreads();
  float bv = bias[f];
  for (int t = tid; t < Lc; t += 256) {
    float y = bv;
    for (int k = 0; k < K; ++k) y = fmaf(xs[t + k], wsm[k], y);
    yy[t] = y * y;
  }
  __syncthreads();
  for (int lp = tid; lp < Lp; lp += 256) {
    float s = 0.f;
    #pragma unroll
    for (int p = 0; p < 16; ++p) s += yy[lp * 4 + p];
    float val = __logf(s * (1.f / 16.f));
    int l0g  = off + lp;
    int flat = (f * 32 + c) * LTOT + l0g;     // within-batch flat index
    int c2   = flat / 35008;
    int j    = flat % 35008;
    int l    = j >> 6;
    int kcol = c2 * 64 + (j & 63);
    X16[((size_t)b * LTOT + l) * KFI + kcol] = (_Float16)val;
  }
}

// ---------------------------------------------------------------------------
// f16 WMMA fragment loader, ISA 16-bit A 16x32 layout (row-major source).
// ---------------------------------------------------------------------------
__device__ inline v16h frag16(const _Float16* __restrict__ p, int ld, int row, int kk, int lane) {
  union { v16h v; unsigned u[8]; } r;
  const _Float16* q = p + (size_t)(row + (lane & 15)) * ld + kk + ((lane >> 4) << 3);
  #pragma unroll
  for (int v = 0; v < 8; ++v) {
    int k = ((v & 4) << 2) + ((v & 3) << 1);   // (v>=4)*16 + 2*(v%4)
    r.u[v] = *(const unsigned*)(q + k);
  }
  return r.v;
}

// C = A(MxK f16) * Bt(NxK f16)^T + bias. Each wave: 16x32 output (A reuse).
// Requires K%32==0, N%32==0.
__global__ __launch_bounds__(128) void k_gemm(const _Float16* __restrict__ A,
                                              const _Float16* __restrict__ Bt,
                                              const float* __restrict__ bias,
                                              float* __restrict__ C,
                                              int M, int N, int K) {
  int lane  = threadIdx.x & 31;
  int wave  = threadIdx.x >> 5;
  int tilesN2 = N >> 5;
  int tile  = blockIdx.x * 4 + wave;
  int tm = tile / tilesN2, tn = (tile % tilesN2) << 5;
  int row0 = tm << 4;
  if (row0 >= M) return;
  v8f acc0 = {}, acc1 = {};
  for (int kk = 0; kk < K; kk += 32) {
    if (kk + 64 <= K)
      __builtin_prefetch(A + (size_t)(row0 + (lane & 15)) * K + kk + 32, 0, 0);
    v16h a  = frag16(A, K, row0, kk, lane);
    v16h b0 = frag16(Bt, K, tn, kk, lane);
    v16h b1 = frag16(Bt, K, tn + 16, kk, lane);
    acc0 = __builtin_amdgcn_wmma_f32_16x16x32_f16(false, a, false, b0, (short)0, acc0, false, false);
    acc1 = __builtin_amdgcn_wmma_f32_16x16x32_f16(false, a, false, b1, (short)0, acc1, false, false);
  }
  int n0 = tn + (lane & 15);
  float bv0 = bias ? bias[n0] : 0.f;
  float bv1 = bias ? bias[n0 + 16] : 0.f;
  int mBase = row0 + ((lane >> 4) << 3);
  #pragma unroll
  for (int r = 0; r < 8; ++r) {
    size_t rb = (size_t)(mBase + r) * N;
    C[rb + n0]      = acc0[r] + bv0;
    C[rb + n0 + 16] = acc1[r] + bv1;
  }
}

// ---------------------------------------------------------------------------
// TDM-staged GEMM for the big fi operand (N==32 fixed). Each wave DMAs its
// 16x64-half A band into LDS via tensor_load_to_lds (2-D descriptor,
// tensor_dim1 = remaining rows so hardware zero-fills the M tail), double-
// buffered with s_wait_tensorcnt, then feeds WMMA from LDS.
// ---------------------------------------------------------------------------
__global__ __launch_bounds__(128) void k_gemm_tdm(const _Float16* __restrict__ A,
                                                  const _Float16* __restrict__ Bt,
                                                  const float* __restrict__ bias,
                                                  float* __restrict__ C,
                                                  int M, int K) {
  __shared__ _Float16 sA[4][2][16 * 64];   // 4 waves x double buffer x 16x64 halves
  int lane = threadIdx.x & 31;
  int wave = threadIdx.x >> 5;
  int row0 = blockIdx.x * 64 + wave * 16;
  if (row0 >= M) return;
  unsigned lds0 = (unsigned)(size_t)(&sA[wave][0][0]);
  unsigned lds1 = (unsigned)(size_t)(&sA[wave][1][0]);
  unsigned td1 = (unsigned)(M - row0);     // rows remaining -> OOB rows read 0

  auto issue = [&](int kk, unsigned laddr) {
    unsigned long long ga = (unsigned long long)(size_t)(A + (size_t)row0 * K + kk);
    u32x4 g0;
    g0[0] = 1u;                                            // count=1, user mode
    g0[1] = laddr;                                         // lds_addr
    g0[2] = (unsigned)(ga & 0xffffffffu);                  // global_addr lo
    g0[3] = (unsigned)((ga >> 32) & 0x01ffffffu) | 0x80000000u;  // addr hi | type=2
    u32x8 g1;
    g1[0] = 0x00010000u;                                   // data_size = 2 bytes
    g1[1] = ((unsigned)K & 0xffffu) << 16;                 // tensor_dim0 lo
    g1[2] = (((unsigned)K >> 16) & 0xffffu) | ((td1 & 0xffffu) << 16);  // td0 hi | td1 lo
    g1[3] = ((td1 >> 16) & 0xffffu) | (64u << 16);         // td1 hi | tile_dim0=64
    g1[4] = 16u;                                           // tile_dim1=16, tile_dim2=0
    g1[5] = (unsigned)K;                                   // tensor_dim0_stride lo
    g1[6] = 0u;                                            // stride hi | dim1_stride lo
    g1[7] = 0u;
    asm volatile("tensor_load_to_lds %0, %1" :: "s"(g0), "s"(g1) : "memory");
  };

  v8f acc0 = {}, acc1 = {};
  issue(0, lds0);
  int nChunks = K >> 6;                                    // K % 64 == 0
  for (int ch = 0; ch < nChunks; ++ch) {
    if (ch + 1 < nChunks) {
      issue((ch + 1) << 6, (ch & 1) ? lds0 : lds1);
      __builtin_amdgcn_s_wait_tensorcnt(1);                // chunk ch complete (in-order)
    } else {
      __builtin_amdgcn_s_wait_tensorcnt(0);
    }
    const _Float16* Abuf = &sA[wave][ch & 1][0];
    #pragma unroll
    for (int s = 0; s < 2; ++s) {
      int kk = (ch << 6) + s * 32;
      v16h a  = frag16(Abuf, 64, 0, s * 32, lane);
      v16h b0 = frag16(Bt, K, 0, kk, lane);
      v16h b1 = frag16(Bt, K, 16, kk, lane);
      acc0 = __builtin_amdgcn_wmma_f32_16x16x32_f16(false, a, false, b0, (short)0, acc0, false, false);
      acc1 = __builtin_amdgcn_wmma_f32_16x16x32_f16(false, a, false, b1, (short)0, acc1, false, false);
    }
  }
  int n = lane & 15;
  float bv0 = bias[n], bv1 = bias[n + 16];
  int mBase = row0 + ((lane >> 4) << 3);
  #pragma unroll
  for (int r = 0; r < 8; ++r) {
    int m = mBase + r;
    if (m < M) {
      C[(size_t)m * 32 + n]      = acc0[r] + bv0;
      C[(size_t)m * 32 + n + 16] = acc1[r] + bv1;
    }
  }
}

// ---------------------------------------------------------------------------
// dtype conversions / packing
// ---------------------------------------------------------------------------
__global__ void k_f32_to_f16(const float* __restrict__ s, _Float16* __restrict__ d, int n) {
  int i = blockIdx.x * blockDim.x + threadIdx.x;
  if (i < n) d[i] = (_Float16)s[i];
}
__global__ void k_pack_pad(const float* __restrict__ s, _Float16* __restrict__ d,
                           int rows, int K, int Kp) {
  int i = blockIdx.x * blockDim.x + threadIdx.x;
  if (i >= rows * Kp) return;
  int r = i / Kp, k = i % Kp;
  d[i] = (k < K) ? (_Float16)s[(size_t)r * K + k] : (_Float16)0.f;
}
__global__ void k_transpose_pad(const float* __restrict__ s /*KxN*/, _Float16* __restrict__ d /*NxKp*/,
                                int K, int N, int Kp) {
  int i = blockIdx.x * blockDim.x + threadIdx.x;
  if (i >= N * Kp) return;
  int n = i / Kp, k = i % Kp;
  d[i] = (k < K) ? (_Float16)s[(size_t)k * N + n] : (_Float16)0.f;
}

// ---------------------------------------------------------------------------
// LayerNorm over full (100,32) window; window is a contiguous 3200-slab of fi.
// ---------------------------------------------------------------------------
__global__ __launch_bounds__(128) void k_ln1(const float* __restrict__ fi,
                                             const float* __restrict__ g,
                                             const float* __restrict__ bo,
                                             float* __restrict__ h, _Float16* __restrict__ h16) {
  int seq = blockIdx.x;
  int b = seq / NW, w = seq % NW;
  const float* base = fi + ((size_t)b * LTOT + w * 20) * 32;
  __shared__ float red[128];
  int tid = threadIdx.x;
  float s = 0.f;
  for (int i = tid; i < 3200; i += 128) s += base[i];
  red[tid] = s; __syncthreads();
  for (int st = 64; st > 0; st >>= 1) { if (tid < st) red[tid] += red[tid + st]; __syncthreads(); }
  float mu = red[0] * (1.f / 3200.f);
  __syncthreads();
  float v = 0.f;
  for (int i = tid; i < 3200; i += 128) { float d = base[i] - mu; v = fmaf(d, d, v); }
  red[tid] = v; __syncthreads();
  for (int st = 64; st > 0; st >>= 1) { if (tid < st) red[tid] += red[tid + st]; __syncthreads(); }
  float inv = rsqrtf(red[0] * (1.f / 3200.f) + EPSF);
  for (int i = tid; i < 3200; i += 128) {
    float val = (base[i] - mu) * inv * g[i] + bo[i];
    size_t o = (size_t)seq * 3200 + i;
    h[o] = val; h16[o] = (_Float16)val;
  }
}

// ---------------------------------------------------------------------------
// Attention core: per (seq, head). head_dim=4 == WMMA f32 K, S=100 -> 7 tiles.
// ---------------------------------------------------------------------------
__global__ __launch_bounds__(224) void k_attn(const float* __restrict__ qkv,
                                              _Float16* __restrict__ o16) {
  int seq = blockIdx.x, head = blockIdx.y;
  int lane = threadIdx.x & 31;
  int wave = threadIdx.x >> 5;          // 0..6, 16 query rows each
  __shared__ float attn_s[7][16][104];
  const float* base = qkv + (size_t)seq * WINL * 96;
  const float scale = 0.5f;             // 1/sqrt(4)
  int row0 = wave * 16;
  int mq = row0 + (lane & 15);
  int k0 = (lane >> 4) * 2;
  bool mv = mq < WINL;
  v2f af;
  af[0] = mv ? base[(size_t)mq * 96 + head * 4 + k0] * scale : 0.f;
  af[1] = mv ? base[(size_t)mq * 96 + head * 4 + k0 + 1] * scale : 0.f;
  v8f sc[7];
  #pragma unroll
  for (int j = 0; j < 7; ++j) {
    int nk = j * 16 + (lane & 15);
    bool nv = nk < WINL;
    v2f bf;
    bf[0] = nv ? base[(size_t)nk * 96 + 32 + head * 4 + k0] : 0.f;
    bf[1] = nv ? base[(size_t)nk * 96 + 32 + head * 4 + k0 + 1] : 0.f;
    v8f z = {};
    sc[j] = __builtin_amdgcn_wmma_f32_16x16x4_f32(false, af, false, bf, (short)0, z, false, false);
  }
  if ((lane & 15) + 96 >= WINL) {
    #pragma unroll
    for (int r = 0; r < 8; ++r) sc[6][r] = -1e30f;
  }
  #pragma unroll
  for (int r = 0; r < 8; ++r) {
    float mx = -1e30f;
    #pragma unroll
    for (int j = 0; j < 7; ++j) mx = fmaxf(mx, sc[j][r]);
    for (int o = 1; o < 16; o <<= 1) mx = fmaxf(mx, __shfl_xor(mx, o, 32));
    float sum = 0.f;
    #pragma unroll
    for (int j = 0; j < 7; ++j) { float e = __expf(sc[j][r] - mx); sc[j][r] = e; sum += e; }
    for (int o = 1; o < 16; o <<= 1) sum += __shfl_xor(sum, o, 32);
    float inv = 1.f / sum;
    #pragma unroll
    for (int j = 0; j < 7; ++j) sc[j][r] *= inv;
  }
  #pragma unroll
  for (int j = 0; j < 7; ++j) {
    int n = j * 16 + (lane & 15);
    if (n < WINL) {
      #pragma unroll
      for (int r = 0; r < 8; ++r)
        attn_s[wave][r + ((lane >> 4) << 3)][n] = sc[j][r];
    }
  }
  __syncthreads();
  v8f oc = {};
  int ml = lane & 15;
  for (int s = 0; s < 25; ++s) {
    int kb = s * 4 + k0;
    v2f a2; a2[0] = attn_s[wave][ml][kb]; a2[1] = attn_s[wave][ml][kb + 1];
    v2f b2;
    bool nv = ml < 4;
    b2[0] = nv ? base[(size_t)kb * 96 + 64 + head * 4 + ml] : 0.f;
    b2[1] = nv ? base[(size_t)(kb + 1) * 96 + 64 + head * 4 + ml] : 0.f;
    oc = __builtin_amdgcn_wmma_f32_16x16x4_f32(false, a2, false, b2, (short)0, oc, false, false);
  }
  #pragma unroll
  for (int r = 0; r < 8; ++r) {
    int M = row0 + r + ((lane >> 4) << 3);
    int n = lane & 15;
    if (M < WINL && n < 4)
      o16[((size_t)seq * WINL + M) * 32 + head * 4 + n] = (_Float16)oc[r];
  }
}

// residual + LayerNorm over last dim (32): one wave per row
__global__ __launch_bounds__(128) void k_ln2(const float* __restrict__ ap, const float* __restrict__ h,
                                             const float* __restrict__ g, const float* __restrict__ bo,
                                             float* __restrict__ out) {
  int row = blockIdx.x * 4 + (threadIdx.x >> 5);
  int lane = threadIdx.x & 31;
  size_t idx = (size_t)row * 32 + lane;
  float v = ap[idx] + h[idx];
  float mu = v;
  for (int o = 1; o < 32; o <<= 1) mu += __shfl_xor(mu, o, 32);
  mu *= (1.f / 32.f);
  float d = v - mu, va = d * d;
  for (int o = 1; o < 32; o <<= 1) va += __shfl_xor(va, o, 32);
  va *= (1.f / 32.f);
  out[idx] = d * rsqrtf(va + EPSF) * g[lane] + bo[lane];
}

// TCN: conv1d(k=3, pad=1) over time per window, + BN(eval) + relu
__global__ void k_tcn(const float* __restrict__ a, const float* __restrict__ w,
                      const float* __restrict__ bias, const float* __restrict__ bng,
                      const float* __restrict__ bnb, float* __restrict__ tbuf) {
  int gid = blockIdx.x * blockDim.x + threadIdx.x;
  if (gid >= SEQS * 32 * WINL) return;
  int t = gid % WINL; int r = gid / WINL;
  int oc = r % 32; int seq = r / 32;
  float y = bias[oc];
  for (int ic = 0; ic < 32; ++ic)
    #pragma unroll
    for (int kk = 0; kk < 3; ++kk) {
      int tt = t + kk - 1;
      if (tt >= 0 && tt < WINL)
        y = fmaf(a[((size_t)seq * WINL + tt) * 32 + ic], w[(oc * 32 + ic) * 3 + kk], y);
    }
  y = bng[oc] * y * rsqrtf(1.f + EPSF) + bnb[oc];
  tbuf[gid] = fmaxf(y, 0.f);
}

// reshape/transpose scramble -> stacked (B, 32, 2300)
__global__ void k_stack(const float* __restrict__ tbuf, float* __restrict__ st) {
  int gid = blockIdx.x * blockDim.x + threadIdx.x;
  if (gid >= BATCH * 32 * 2300) return;
  int col = gid % 2300; int r = gid / 2300;
  int row = r % 32; int b = r / 32;
  int flat = row * 2300 + col;            // (WIN,32,NW) flattened
  int wt = flat / (32 * NW); int rem = flat % (32 * NW);
  int c = rem / NW; int w = rem % NW;
  st[gid] = tbuf[(((size_t)(b * NW + w)) * 32 + c) * WINL + wt];
}

// fusion conv1d (k=3, pad=1)
__global__ void k_fusion(const float* __restrict__ st, const float* __restrict__ w,
                         const float* __restrict__ bias, float* __restrict__ fu) {
  int gid = blockIdx.x * blockDim.x + threadIdx.x;
  if (gid >= BATCH * 32 * 2300) return;
  int col = gid % 2300; int r = gid / 2300;
  int oc = r % 32; int b = r / 32;
  float y = bias[oc];
  for (int ic = 0; ic < 32; ++ic)
    #pragma unroll
    for (int kk = 0; kk < 3; ++kk) {
      int cc = col + kk - 1;
      if (cc >= 0 && cc < 2300)
        y = fmaf(st[((size_t)b * 32 + ic) * 2300 + cc], w[(oc * 32 + ic) * 3 + kk], y);
    }
  fu[gid] = y;
}

// relu(fused*lf_w - lf_b), mean over 4 ch per area, BN1(eval)
__global__ void k_lf(const float* __restrict__ fu, const float* __restrict__ lfw,
                     const float* __restrict__ lfb, const float* __restrict__ bg,
                     const float* __restrict__ bb, float* __restrict__ xg) {
  int gid = blockIdx.x * blockDim.x + threadIdx.x;
  if (gid >= BATCH * 8 * 2300) return;
  int col = gid % 2300; int r = gid / 2300;
  int area = r % 8; int b = r / 8;
  float s = 0.f;
  #pragma unroll
  for (int cc = 0; cc < 4; ++cc) {
    int ch = area * 4 + cc;
    float v = fu[((size_t)b * 32 + ch) * 2300 + col] * lfw[(size_t)ch * 2300 + col] - lfb[ch];
    s += fmaxf(v, 0.f);
  }
  s *= 0.25f;
  xg[gid] = bg[area] * s * rsqrtf(1.f + EPSF) + bb[area];
}

// dynamic GCN adjacency: s = x x^T (8x8), normalize, out = relu(adj @ h)
__global__ __launch_bounds__(256) void k_gcn_adj(const float* __restrict__ x,
                                                 const float* __restrict__ h,
                                                 float* __restrict__ xo, int K, int N) {
  int b = blockIdx.x;
  __shared__ float sA[8][8];
  __shared__ float dS[8];
  const float* xb = x + (size_t)b * 8 * K;
  int tid = threadIdx.x;
  if (tid < 64) {
    int n = tid >> 3, m = tid & 7;
    float s = 0.f;
    for (int k = 0; k < K; ++k) s = fmaf(xb[(size_t)n * K + k], xb[(size_t)m * K + k], s);
    if (n == m) s += 1.f;
    sA[n][m] = s;
  }
  __syncthreads();
  if (tid < 8) {
    float rs = 0.f;
    for (int m = 0; m < 8; ++m) rs += sA[tid][m];
    if (rs == 0.f) rs = 1.f;
    dS[tid] = rsqrtf(rs);
  }
  __syncthreads();
  const float* hb = h + (size_t)b * 8 * N;
  float* ob = xo + (size_t)b * 8 * N;
  for (int idx = tid; idx < 8 * N; idx += 256) {
    int n = idx / N, f = idx % N;
    float acc = 0.f;
    #pragma unroll
    for (int m = 0; m < 8; ++m)
      acc = fmaf(sA[n][m] * dS[n] * dS[m], hb[(size_t)m * N + f], acc);
    ob[idx] = fmaxf(acc, 0.f);
  }
}

// BN2 + final FC (256 -> 4): one block, 128 threads = 32 batches x 4 outputs
__global__ __launch_bounds__(128) void k_final(const float* __restrict__ x3,
                                               const float* __restrict__ bg, const float* __restrict__ bb,
                                               const float* __restrict__ fcw, const float* __restrict__ fcb,
                                               float* __restrict__ out) {
  int tid = threadIdx.x;
  int b = tid >> 2, j = tid & 3;
  float bs = rsqrtf(1.f + EPSF);
  float acc = fcb[j];
  for (int i = 0; i < 256; ++i) {
    int n = i >> 5;
    float v = bg[n] * x3[(size_t)b * 256 + i] * bs + bb[n];
    acc = fmaf(v, fcw[i * 4 + j], acc);
  }
  out[b * 4 + j] = acc;
}

// ---------------------------------------------------------------------------
extern "C" void kernel_launch(void* const* d_in, const int* in_sizes, int n_in,
                              void* d_out, int out_size, void* d_ws, size_t ws_size,
                              hipStream_t stream) {
  const float* x       = (const float*)d_in[0];
  const float* tc1_w   = (const float*)d_in[1];  const float* tc1_b = (const float*)d_in[2];
  const float* tc2_w   = (const float*)d_in[3];  const float* tc2_b = (const float*)d_in[4];
  const float* tc3_w   = (const float*)d_in[5];  const float* tc3_b = (const float*)d_in[6];
  const float* fi_w    = (const float*)d_in[7];  const float* fi_b  = (const float*)d_in[8];
  const float* ln1_g   = (const float*)d_in[9];  const float* ln1_b = (const float*)d_in[10];
  const float* qkv_w   = (const float*)d_in[11]; const float* qkv_b = (const float*)d_in[12];
  const float* attn_ow = (const float*)d_in[13]; const float* attn_ob = (const float*)d_in[14];
  const float* ln2_g   = (const float*)d_in[15]; const float* ln2_b = (const float*)d_in[16];
  const float* tcn_w   = (const float*)d_in[17]; const float* tcn_b = (const float*)d_in[18];
  const float* tbng    = (const float*)d_in[19]; const float* tbnb  = (const float*)d_in[20];
  const float* fus_w   = (const float*)d_in[21]; const float* fus_b = (const float*)d_in[22];
  const float* lf_w    = (const float*)d_in[23]; const float* lf_b  = (const float*)d_in[24];
  const float* bn1_g   = (const float*)d_in[25]; const float* bn1_b = (const float*)d_in[26];
  const float* g1_w    = (const float*)d_in[27]; const float* g1_b  = (const float*)d_in[28];
  const float* g2_w    = (const float*)d_in[29]; const float* g2_b  = (const float*)d_in[30];
  const float* g3_w    = (const float*)d_in[31]; const float* g3_b  = (const float*)d_in[32];
  const float* bn2_g   = (const float*)d_in[33]; const float* bn2_b = (const float*)d_in[34];
  const float* fc_w    = (const float*)d_in[35]; const float* fc_b  = (const float*)d_in[36];
  float* out = (float*)d_out;

  char* ws = (char*)d_ws; size_t off = 0;
  auto alloc = [&](size_t bytes) -> void* {
    void* p = ws + off; off += (bytes + 255) & ~(size_t)255; return p;
  };
  _Float16* X16   = (_Float16*)alloc((size_t)MFI * KFI * 2);
  _Float16* wfi16 = (_Float16*)alloc(32 * 2048 * 2);
  _Float16* wqkv16= (_Float16*)alloc(96 * 32 * 2);
  _Float16* wow16 = (_Float16*)alloc(32 * 32 * 2);
  float*    fiout = (float*)alloc((size_t)MFI * 32 * 4);
  float*    h     = (float*)alloc((size_t)ROWS * 32 * 4);
  _Float16* h16   = (_Float16*)alloc((size_t)ROWS * 32 * 2);
  float*    qkvb  = (float*)alloc((size_t)ROWS * 96 * 4);
  _Float16* o16   = (_Float16*)alloc((size_t)ROWS * 32 * 2);
  float*    aproj = (float*)alloc((size_t)ROWS * 32 * 4);
  float*    xg    = (float*)alloc((size_t)BATCH * 8 * 2300 * 4);
  _Float16* x16b  = (_Float16*)alloc((size_t)256 * 2304 * 2);
  _Float16* wT16  = (_Float16*)alloc((size_t)800 * 2304 * 2);
  float*    hbuf  = (float*)alloc((size_t)256 * 800 * 4);
  float*    xg2   = (float*)alloc((size_t)256 * 800 * 4);
  float*    xg3   = (float*)alloc((size_t)256 * 800 * 4);
  float*    xfin  = (float*)alloc((size_t)256 * 32 * 4);
  // aliases into the (large, then-dead) qkv / h / aproj regions:
  float* abuf    = qkvb;                           // post-LN2 activations
  float* stacked = qkvb + (size_t)ROWS * 24;       // second part of qkv region
  float* tbuf    = h;                              // TCN output (h dead after LN2)
  float* fused   = aproj;                          // fusion output (aproj dead)

  auto cdiv = [](long a, long b) { return (int)((a + b - 1) / b); };

  // 1) fused conv+power+scatter (three temporal scales), one block per row
  k_conv_power<<<BATCH*64*32, 256, 0, stream>>>(x, tc1_w, tc1_b, X16, 100, 701, 172, 0);
  k_conv_power<<<BATCH*64*32, 256, 0, stream>>>(x, tc2_w, tc2_b, X16,  50, 751, 184, 172);
  k_conv_power<<<BATCH*64*32, 256, 0, stream>>>(x, tc3_w, tc3_b, X16,  25, 776, 191, 356);

  // 2) weight conversions
  k_f32_to_f16<<<cdiv(32*2048, 256), 256, 0, stream>>>(fi_w, wfi16, 32*2048);
  k_f32_to_f16<<<cdiv(96*32, 256), 256, 0, stream>>>(qkv_w, wqkv16, 96*32);
  k_f32_to_f16<<<cdiv(32*32, 256), 256, 0, stream>>>(attn_ow, wow16, 32*32);

  // 3) fi conv-as-GEMM via TDM double-buffered staging
  k_gemm_tdm<<<cdiv(MFI, 64), 128, 0, stream>>>(X16, wfi16, fi_b, fiout, MFI, KFI);

  // 4) window LayerNorm
  k_ln1<<<SEQS, 128, 0, stream>>>(fiout, ln1_g, ln1_b, h, h16);

  // 5) QKV GEMM: (73600 x 32) @ (96 x 32)^T
  k_gemm<<<cdiv((long)(ROWS/16)*(96/32), 4), 128, 0, stream>>>(h16, wqkv16, qkv_b, qkvb, ROWS, 96, 32);

  // 6) attention (f32 WMMA, head_dim=4)
  {
    dim3 g(SEQS, 8);
    k_attn<<<g, 224, 0, stream>>>(qkvb, o16);
  }

  // 7) output projection GEMM + residual LN
  k_gemm<<<cdiv((long)(ROWS/16)*(32/32), 4), 128, 0, stream>>>(o16, wow16, attn_ob, aproj, ROWS, 32, 32);
  k_ln2<<<ROWS/4, 128, 0, stream>>>(aproj, h, ln2_g, ln2_b, abuf);

  // 8) TCN + BN + relu
  k_tcn<<<cdiv((long)SEQS*32*WINL, 256), 256, 0, stream>>>(abuf, tcn_w, tcn_b, tbng, tbnb, tbuf);

  // 9) stack scramble + fusion conv + lf/mean/BN1
  k_stack <<<cdiv((long)BATCH*32*2300, 256), 256, 0, stream>>>(tbuf, stacked);
  k_fusion<<<cdiv((long)BATCH*32*2300, 256), 256, 0, stream>>>(stacked, fus_w, fus_b, fused);
  k_lf    <<<cdiv((long)BATCH*8*2300, 256), 256, 0, stream>>>(fused, lf_w, lf_b, bn1_g, bn1_b, xg);

  // 10) GCN layer 1: K=2300 (pad 2304), N=800
  k_pack_pad     <<<cdiv(256L*2304, 256), 256, 0, stream>>>(xg, x16b, 256, 2300, 2304);
  k_transpose_pad<<<cdiv(800L*2304, 256), 256, 0, stream>>>(g1_w, wT16, 2300, 800, 2304);
  k_gemm<<<cdiv((256/16)*(800/32), 4), 128, 0, stream>>>(x16b, wT16, g1_b, hbuf, 256, 800, 2304);
  k_gcn_adj<<<BATCH, 256, 0, stream>>>(xg, hbuf, xg2, 2300, 800);

  // 11) GCN layer 2: K=800, N=800
  k_pack_pad     <<<cdiv(256L*800, 256), 256, 0, stream>>>(xg2, x16b, 256, 800, 800);
  k_transpose_pad<<<cdiv(800L*800, 256), 256, 0, stream>>>(g2_w, wT16, 800, 800, 800);
  k_gemm<<<cdiv((256/16)*(800/32), 4), 128, 0, stream>>>(x16b, wT16, g2_b, hbuf, 256, 800, 800);
  k_gcn_adj<<<BATCH, 256, 0, stream>>>(xg2, hbuf, xg3, 800, 800);

  // 12) GCN layer 3: K=800, N=32
  k_pack_pad     <<<cdiv(256L*800, 256), 256, 0, stream>>>(xg3, x16b, 256, 800, 800);
  k_transpose_pad<<<cdiv(32L*800, 256), 256, 0, stream>>>(g3_w, wT16, 800, 32, 800);
  k_gemm<<<cdiv((256/16)*(32/32), 4), 128, 0, stream>>>(x16b, wT16, g3_b, hbuf, 256, 32, 800);
  k_gcn_adj<<<BATCH, 256, 0, stream>>>(xg3, hbuf, xfin, 800, 32);

  // 13) BN2 + FC
  k_final<<<1, 128, 0, stream>>>(xfin, bn2_g, bn2_b, fc_w, fc_b, out);

  (void)in_sizes; (void)n_in; (void)out_size; (void)ws_size;
}
